// SpatialTemporalAttnBlock_74929999446753
// MI455X (gfx1250) — matrix operand order repaired and checked by reading, not compiled, not run on hardware
//
#include <hip/hip_runtime.h>
#include <math.h>

typedef __bf16 bf16;
typedef __attribute__((ext_vector_type(16))) __bf16 v16bf;
typedef __attribute__((ext_vector_type(8)))  __bf16 v8bf;
typedef __attribute__((ext_vector_type(8)))  float  v8f;
typedef __attribute__((ext_vector_type(4)))  int    v4i;

#define NB   4
#define DCH  256
#define KTOK 8192   // h*w*t  (t fastest)
#define QTOK 1024   // h*w (t2 == 1)

// gfx1250 async global->LDS copy (ASYNCcnt-tracked), guarded for portability
#if defined(__has_builtin)
#  if __has_builtin(__builtin_amdgcn_global_load_async_to_lds_b128)
#    define HAVE_ASYNC_LDS 1
#  endif
#endif
#ifndef HAVE_ASYNC_LDS
#  define HAVE_ASYNC_LDS 0
#endif

#define GAS __attribute__((address_space(1)))
#define LAS __attribute__((address_space(3)))

// ---------------------------------------------------------------------------
// Generic batched bf16-WMMA GEMM: C[M,N] = A[M,K] x B[K,N] (row major).
// A_F32/B_F32: input stored as f32, converted to bf16 on LDS fill.
// C_BF16: store result as bf16 instead of f32. LEAKY: LeakyReLU(0.2).
// Block tile 64(M) x 128(N), BK=32; 256 threads = 8 waves, each wave 32x32.
// B tile is staged TRANSPOSED in LDS so every WMMA fragment is a contiguous
// 32B LDS read (ds_load_b128 pairs, no scalar ds_load_u16 gathers).
// All dims are exact tile multiples at every launch site -> EXEC all-ones
// around every v_wmma (ISA requirement).
// ---------------------------------------------------------------------------
template<bool A_F32, bool B_F32, bool LEAKY, bool C_BF16>
__global__ __launch_bounds__(256)
void gemm_wmma(const void* __restrict__ Ap, const void* __restrict__ Bp,
               void* __restrict__ Cp, int K, int lda, int ldb, int ldc,
               long long aStride, long long bStride, long long cStride)
{
  constexpr int BM = 64, BN = 128, BK = 32, LDW = BK + 16;   // 48 halves = 96B rows
  __shared__ alignas(16) bf16 As [BM][LDW];   // row-major A tile
  __shared__ alignas(16) bf16 Bst[BN][LDW];   // TRANSPOSED B tile: Bst[n][k]

  const int tid  = threadIdx.x;
  const int lane = tid & 31;
  const int wave = tid >> 5;
  const int wm   = wave >> 2;           // 0..1
  const int wn   = wave & 3;            // 0..3
  const int gm   = blockIdx.y * BM;
  const int gn   = blockIdx.x * BN;

  const float* Af = (const float*)Ap;
  const bf16*  Ab = (const bf16*)Ap;
  const float* Bf = (const float*)Bp;
  const bf16*  Bb = (const bf16*)Bp;
  const long long aOff = aStride * (long long)blockIdx.z;
  const long long bOff = bStride * (long long)blockIdx.z;

  // fill ownership: A -> (row, 8-wide k chunk); B -> (column, 16-wide k half)
  const int arow = tid >> 2;            // 0..63
  const int akg  = (tid & 3) * 8;       // 0,8,16,24
  const int bcol = tid & 127;           // 0..127
  const int bkh  = (tid >> 7) * 16;     // 0 or 16

  v8f acc[2][2];
#pragma unroll
  for (int tm = 0; tm < 2; ++tm)
#pragma unroll
    for (int tn = 0; tn < 2; ++tn)
#pragma unroll
      for (int r = 0; r < 8; ++r) acc[tm][tn][r] = 0.0f;

  const int mrow = lane & 15;
  const int half = lane >> 4;

  for (int k0 = 0; k0 < K; k0 += BK) {
    // ---- A tile fill (64x32) ----
    if constexpr (A_F32) {
      const float* src = Af + aOff + (long long)(gm + arow) * lda + (k0 + akg);
#pragma unroll
      for (int t = 0; t < 8; ++t) As[arow][akg + t] = (bf16)src[t];
    } else {
      const bf16* src = Ab + aOff + (long long)(gm + arow) * lda + (k0 + akg);
#if HAVE_ASYNC_LDS
      __builtin_amdgcn_global_load_async_to_lds_b128(
          (GAS v4i*)(const v4i*)src, (LAS v4i*)&As[arow][akg], 0, 0);
#else
      *(v8bf*)&As[arow][akg] = *(const v8bf*)src;
#endif
    }

    // ---- B tile fill (32x128), transposed into Bst[col][k] ----
    {
      v16bf tmp;
#pragma unroll
      for (int t = 0; t < 16; ++t) {
        long long gi = bOff + (long long)(k0 + bkh + t) * ldb + (gn + bcol);
        tmp[t] = B_F32 ? (bf16)Bf[gi] : Bb[gi];
      }
      *(v16bf*)&Bst[bcol][bkh] = tmp;          // one 32B LDS store
    }

    if constexpr (!A_F32) {
#if HAVE_ASYNC_LDS
      asm volatile("s_wait_asynccnt 0" ::: "memory");
#endif
    }
    __syncthreads();

    // ---- fragments (contiguous k = 16*half + s, same order for A and B) ----
    v16bf afrag[2], bfrag[2];
#pragma unroll
    for (int tm = 0; tm < 2; ++tm) {
      const int rb = wm * 32 + tm * 16 + mrow;
      v8bf lo = *(const v8bf*)&As[rb][16 * half];       // K = 16*half .. +7
      v8bf hi = *(const v8bf*)&As[rb][16 * half + 8];   // K = 16*half+8 .. +15
#pragma unroll
      for (int i = 0; i < 8; ++i) { afrag[tm][i] = lo[i]; afrag[tm][8 + i] = hi[i]; }
    }
#pragma unroll
    for (int tn = 0; tn < 2; ++tn) {
      const int cb = wn * 32 + tn * 16 + mrow;          // lane&15 = column
      bfrag[tn] = *(const v16bf*)&Bst[cb][half * 16];   // contiguous 32B
    }

#pragma unroll
    for (int tm = 0; tm < 2; ++tm)
#pragma unroll
      for (int tn = 0; tn < 2; ++tn)
        acc[tm][tn] = __builtin_amdgcn_wmma_f32_16x16x32_bf16(
            false, afrag[tm], false, bfrag[tn], (short)0, acc[tm][tn],
            false, false);
    __syncthreads();
  }

  // C/D 16x16 f32 layout: VGPR r, lanes 0-15 -> M=r, lanes 16-31 -> M=8+r
  float* Cf = (float*)Cp + (C_BF16 ? 0 : cStride * (long long)blockIdx.z);
  bf16*  Cb = (bf16*)Cp  + (C_BF16 ? cStride * (long long)blockIdx.z : 0);
#pragma unroll
  for (int tm = 0; tm < 2; ++tm)
#pragma unroll
    for (int tn = 0; tn < 2; ++tn)
#pragma unroll
      for (int r = 0; r < 8; ++r) {
        int row = gm + wm * 32 + tm * 16 + half * 8 + r;
        int col = gn + wn * 32 + tn * 16 + mrow;
        float x = acc[tm][tn][r];
        if (LEAKY) x = (x >= 0.0f) ? x : 0.2f * x;
        long long ci = (long long)row * ldc + col;
        if constexpr (C_BF16) Cb[ci] = (bf16)x; else Cf[ci] = x;
      }
}

// ---------------------------------------------------------------------------
__global__ void f32_to_bf16(const float* __restrict__ in, bf16* __restrict__ out, int n) {
  int i = blockIdx.x * blockDim.x + threadIdx.x;
  if (i < n) out[i] = (bf16)in[i];
}

// vB[n][c][k], k = hw*8 + tt  <-  v[n][tt][c][hw]
__global__ void repack_v(const float* __restrict__ v, bf16* __restrict__ vB) {
  int idx = blockIdx.x * blockDim.x + threadIdx.x;     // 4*256*8192 total
  int k = idx & (KTOK - 1);
  int c = (idx >> 13) & 255;
  int n = idx >> 21;
  int hw = k >> 3, tt = k & 7;
  vB[idx] = (bf16)v[(((long long)(n * 8 + tt) * DCH + c) << 10) + hw];
}

// one block per (n,q) column: max & sum of exp over K=8192 (scores recomputed)
__global__ __launch_bounds__(256)
void softmax_stats(const float* __restrict__ kt, const float* __restrict__ qt,
                   float* __restrict__ mArr, float* __restrict__ sArr) {
  __shared__ float red[256];
  const int n = blockIdx.x >> 10;
  const int q = blockIdx.x & 1023;
  const float q0 = qt[(n * 3 + 0) * 1024 + q];
  const float q1 = qt[(n * 3 + 1) * 1024 + q];
  const float q2 = qt[(n * 3 + 2) * 1024 + q];
  const float sc = 0.57735026918962576f;   // 1/sqrt(3)

  float lmax = -3.4e38f;
  for (int k = threadIdx.x; k < KTOK; k += 256) {
    int hw = k >> 3, tt = k & 7;
    int base = ((n * 8 + tt) * 3) * 1024 + hw;
    float s = (kt[base] * q0 + kt[base + 1024] * q1 + kt[base + 2048] * q2) * sc;
    lmax = fmaxf(lmax, s);
  }
  red[threadIdx.x] = lmax;
  __syncthreads();
  for (int off = 128; off > 0; off >>= 1) {
    if (threadIdx.x < off) red[threadIdx.x] = fmaxf(red[threadIdx.x], red[threadIdx.x + off]);
    __syncthreads();
  }
  const float gmax = red[0];
  __syncthreads();

  float lsum = 0.0f;
  for (int k = threadIdx.x; k < KTOK; k += 256) {
    int hw = k >> 3, tt = k & 7;
    int base = ((n * 8 + tt) * 3) * 1024 + hw;
    float s = (kt[base] * q0 + kt[base + 1024] * q1 + kt[base + 2048] * q2) * sc;
    lsum += __expf(s - gmax);
  }
  red[threadIdx.x] = lsum;
  __syncthreads();
  for (int off = 128; off > 0; off >>= 1) {
    if (threadIdx.x < off) red[threadIdx.x] += red[threadIdx.x + off];
    __syncthreads();
  }
  if (threadIdx.x == 0) { mArr[blockIdx.x] = gmax; sArr[blockIdx.x] = 1.0f / red[0]; }
}

// one block per (n,k) row: coalesced write of attn[n][k][0..1023]
__global__ __launch_bounds__(256)
void attn_write(const float* __restrict__ kt, const float* __restrict__ qt,
                const float* __restrict__ mArr, const float* __restrict__ sArr,
                float* __restrict__ attn) {
  const int n = blockIdx.x >> 13;
  const int k = blockIdx.x & 8191;
  const int hw = k >> 3, tt = k & 7;
  const int base = ((n * 8 + tt) * 3) * 1024 + hw;
  const float k0 = kt[base], k1 = kt[base + 1024], k2 = kt[base + 2048];
  const float sc = 0.57735026918962576f;
  const long long rowOff = ((long long)n * KTOK + k) * QTOK;
  for (int q = threadIdx.x; q < QTOK; q += 256) {
    float s = (k0 * qt[(n * 3 + 0) * 1024 + q] +
               k1 * qt[(n * 3 + 1) * 1024 + q] +
               k2 * qt[(n * 3 + 2) * 1024 + q]) * sc;
    attn[rowOff + q] = __expf(s - mArr[n * 1024 + q]) * sArr[n * 1024 + q];
  }
}

// im2col for 3x3 SAME conv: imB[n][din*9+ky*3+kx][y*32+x]
__global__ void im2col3x3(const float* __restrict__ pv, bf16* __restrict__ imB) {
  long long idx = (long long)blockIdx.x * blockDim.x + threadIdx.x;  // 4*2304*1024
  int pix = (int)(idx & 1023);
  int kk  = (int)((idx >> 10) % 2304);
  int n   = (int)(idx / (2304LL * 1024));
  int y = pix >> 5, x = pix & 31;
  int din = kk / 9, r = kk % 9, ky = r / 3, kx = r % 3;
  int py = y + ky - 1, px = x + kx - 1;
  float val = 0.0f;
  if ((unsigned)py < 32u && (unsigned)px < 32u)
    val = pv[((long long)(n * 256 + din) << 10) + py * 32 + px];
  imB[idx] = (bf16)val;
}

// ---------------------------------------------------------------------------
extern "C" void kernel_launch(void* const* d_in, const int* in_sizes, int n_in,
                              void* d_out, int out_size, void* d_ws, size_t ws_size,
                              hipStream_t stream) {
  const float* kt   = (const float*)d_in[0];
  const float* qt   = (const float*)d_in[1];
  const float* vt   = (const float*)d_in[2];
  const float* Wv   = (const float*)d_in[3];
  const float* Wout = (const float*)d_in[4];
  float* out = (float*)d_out;

  // output regions (flat, return order): v_out, k, q, attn
  float* out_v    = out;
  float* out_k    = out + 1048576;
  float* out_q    = out + 1048576 + 98304;
  float* out_attn = out + 1048576 + 98304 + 12288;   // (n, K, Q) == reshaped attn

  // workspace carve-up (all blocks 16B-aligned)
  char* ws = (char*)d_ws;
  bf16*  vB    = (bf16*)ws;   ws += (size_t)NB * DCH * KTOK * 2;   // 16 MB
  bf16*  valbf = (bf16*)ws;   ws += (size_t)NB * DCH * KTOK * 2;   // 16 MB
  float* mArr  = (float*)ws;  ws += 4096 * 4;
  float* sArr  = (float*)ws;  ws += 4096 * 4;
  float* pval  = (float*)ws;  ws += (size_t)NB * DCH * QTOK * 4;   // 4 MB
  bf16*  imB   = (bf16*)ws;   ws += (size_t)NB * 2304 * 1024 * 2;  // 18 MB
  bf16*  WvB   = (bf16*)ws;   ws += 256 * 256 * 2;
  bf16*  WoutB = (bf16*)ws;   ws += 256 * 2304 * 2;

  // passthrough outputs (d2d async copies are graph-capture safe)
  (void)hipMemcpyAsync(out_k, kt, 98304 * sizeof(float), hipMemcpyDeviceToDevice, stream);
  (void)hipMemcpyAsync(out_q, qt, 12288 * sizeof(float), hipMemcpyDeviceToDevice, stream);

  // weight conversions + v repack
  f32_to_bf16<<<(65536 + 255) / 256, 256, 0, stream>>>(Wv, WvB, 65536);
  f32_to_bf16<<<(589824 + 255) / 256, 256, 0, stream>>>(Wout, WoutB, 589824);
  repack_v<<<8388608 / 256, 256, 0, stream>>>(vt, vB);

  // Stage A: val[n] = W_v x vB[n]   (M=256, N=8192, K=256), bf16 output
  gemm_wmma<false, false, false, true><<<dim3(8192 / 128, 256 / 64, NB), 256, 0, stream>>>(
      WvB, vB, valbf, 256, 256, 8192, 8192,
      0LL, (long long)DCH * KTOK, (long long)DCH * KTOK);

  // Stage B: softmax over keys, attn written once, coalesced
  softmax_stats<<<NB * 1024, 256, 0, stream>>>(kt, qt, mArr, sArr);
  attn_write<<<NB * 8192, 256, 0, stream>>>(kt, qt, mArr, sArr, out_attn);

  // Stage C: pval[n] = val[n] x attn[n]   (M=256, N=1024, K=8192)
  // A: bf16 (async LDS fill), B: f32 attn converted on fill
  gemm_wmma<false, true, false, false><<<dim3(1024 / 128, 256 / 64, NB), 256, 0, stream>>>(
      valbf, out_attn, pval, 8192, 8192, 1024, 1024,
      (long long)DCH * KTOK, (long long)KTOK * QTOK, (long long)DCH * QTOK);

  // Stage D: 3x3 conv as implicit GEMM + LeakyReLU  (M=256, N=1024, K=2304)
  im2col3x3<<<(NB * 2304 * 1024) / 256, 256, 0, stream>>>(pval, imB);
  gemm_wmma<false, false, true, false><<<dim3(1024 / 128, 256 / 64, NB), 256, 0, stream>>>(
      WoutB, imB, out_v, 2304, 2304, 1024, 1024,
      0LL, (long long)2304 * 1024, (long long)DCH * QTOK);
}
